// InterpAttentionKHeadsNet_83528523973199
// MI455X (gfx1250) — matrix operand order repaired and checked by reading, not compiled, not run on hardware
//
#include <hip/hip_runtime.h>
#include <hip/hip_bf16.h>

typedef unsigned short u16;
typedef __attribute__((ext_vector_type(16))) __bf16 v16bf;
typedef __attribute__((ext_vector_type(8)))  float  v8f;

#define LATENT 128
#define NPTS   16384
#define MPTS   32768
#define KNN    16
#define IC1PAD 160   // 131 -> 160 (5 chunks of 32)

// ---------- helpers ----------
__device__ __host__ __forceinline__ u16 f2bf(float f) {
    union { float f; unsigned u; } v; v.f = f;
    unsigned r = v.u + 0x7FFFu + ((v.u >> 16) & 1u);   // round-to-nearest-even
    return (u16)(r >> 16);
}

union FragU { uint4 u[2]; v16bf v; };

// Two 16x16 output tiles sharing the A (weight) fragment: one A load -> two WMMAs.
// Weights: global row-major bf16 [OC][icpad]; activations: LDS [16 cols][istride ch] bf16.
// A-frag (lane L): row M = L%16, halves = K 16*lg + 0..15 (contiguous channels).
// B-frag (lane L): col N = L%16, halves = K 16*lg + 0..15 (contiguous channels).
__device__ __forceinline__ void gemm_tile2(const u16* __restrict__ W, int icpad, int rt,
                                           const u16* in0, const u16* in1, int istride,
                                           int lane, int nchunks, v8f& acc0, v8f& acc1) {
    const int col = lane & 15, lg = lane >> 4;
    const u16* wrow = W + (size_t)(rt * 16 + col) * icpad;
    for (int c = 0; c < nchunks; ++c) {
        FragU a, b0, b1;
        a.u[0] = *reinterpret_cast<const uint4*>(wrow + c * 32 + 16 * lg);
        a.u[1] = *reinterpret_cast<const uint4*>(wrow + c * 32 + 16 * lg + 8);
        const u16* bp0 = in0 + col * istride + c * 32 + 16 * lg;
        const u16* bp1 = in1 + col * istride + c * 32 + 16 * lg;
        b0.u[0] = *reinterpret_cast<const uint4*>(bp0);
        b0.u[1] = *reinterpret_cast<const uint4*>(bp0 + 8);
        b1.u[0] = *reinterpret_cast<const uint4*>(bp1);
        b1.u[1] = *reinterpret_cast<const uint4*>(bp1 + 8);
        acc0 = __builtin_amdgcn_wmma_f32_16x16x32_bf16(false, a.v, false, b0.v,
                                                       (short)0, acc0, false, false);
        acc1 = __builtin_amdgcn_wmma_f32_16x16x32_bf16(false, a.v, false, b1.v,
                                                       (short)0, acc1, false, false);
    }
}

__device__ __forceinline__ void epi_relu_store(const v8f& acc, const float* __restrict__ bias,
                                               int base, u16* dst) {
    unsigned hh[4];
#pragma unroll
    for (int v = 0; v < 8; v += 2) {
        float f0 = fmaxf(acc[v]     + bias[base + v],     0.f);
        float f1 = fmaxf(acc[v + 1] + bias[base + v + 1], 0.f);
        hh[v >> 1] = (unsigned)f2bf(f0) | ((unsigned)f2bf(f1) << 16);
    }
    uint4 pk = {hh[0], hh[1], hh[2], hh[3]};
    *reinterpret_cast<uint4*>(dst) = pk;
}

// ---------- prep: fold BN into bf16 weights + f32 biases ----------
__global__ void prep_weights(
    const float* w1, const float* bi1, const float* g1, const float* be1, const float* m1, const float* v1,
    const float* w2, const float* bi2, const float* g2, const float* be2, const float* m2, const float* v2,
    const float* w3, const float* bi3, const float* g3, const float* be3, const float* m3, const float* v3,
    const float* wq, const float* biq, const float* gq, const float* beq, const float* mq, const float* vq,
    const float* wv, const float* biv, const float* gv, const float* bev, const float* mv, const float* vv,
    u16* W1f, u16* W2f, u16* W3f, u16* Wqf, u16* Wvf,
    float* b1o, float* b2o, float* b3o, float* bqo, float* bvo)
{
    int r = blockIdx.x * blockDim.x + threadIdx.x;
    if (r < 128) {                       // layer 1: [128][131] -> [128][160]
        float s = g1[r] * rsqrtf(v1[r] + 1e-5f);
        for (int c = 0; c < IC1PAD; ++c)
            W1f[r * IC1PAD + c] = f2bf(c < 131 ? s * w1[r * 131 + c] : 0.f);
        b1o[r] = s * (bi1[r] - m1[r]) + be1[r];
    } else if (r < 256) {                // layer 2
        int o = r - 128;
        float s = g2[o] * rsqrtf(v2[o] + 1e-5f);
        for (int c = 0; c < 128; ++c) W2f[o * 128 + c] = f2bf(s * w2[o * 128 + c]);
        b2o[o] = s * (bi2[o] - m2[o]) + be2[o];
    } else if (r < 384) {                // layer 3
        int o = r - 256;
        float s = g3[o] * rsqrtf(v3[o] + 1e-5f);
        for (int c = 0; c < 128; ++c) W3f[o * 128 + c] = f2bf(s * w3[o * 128 + c]);
        b3o[o] = s * (bi3[o] - m3[o]) + be3[o];
    } else if (r < 448) {                // query (64 rows)
        int o = r - 384;
        float s = gq[o] * rsqrtf(vq[o] + 1e-5f);
        for (int c = 0; c < 128; ++c) Wqf[o * 128 + c] = f2bf(s * wq[o * 128 + c]);
        bqo[o] = s * (biq[o] - mq[o]) + beq[o];
    } else if (r < 576) {                // value
        int o = r - 448;
        float s = gv[o] * rsqrtf(vv[o] + 1e-5f);
        for (int c = 0; c < 128; ++c) Wvf[o * 128 + c] = f2bf(s * wv[o * 128 + c]);
        bvo[o] = s * (biv[o] - mv[o]) + bev[o];
    }
}

// ---------- prep: latents [B][128][N] f32 -> latT [B][N][128] bf16 ----------
__global__ void prep_latT(const float* __restrict__ latents, u16* __restrict__ latT, int total) {
    int i = blockIdx.x * blockDim.x + threadIdx.x;
    if (i >= total) return;
    int c = i & 127;
    int n = (i >> 7) & (NPTS - 1);
    int b = i >> (7 + 14);
    latT[i] = f2bf(latents[((size_t)b * 128 + c) * NPTS + n]);
}

// ---------- main: one wave == two (b, m) points, 16 neighbor columns each ----------
struct PtLds {
    u16   bufA[16 * IC1PAD];   // 5120 B  (layer1 input / layer2 output, row stride 160)
    u16   bufB[16 * 128];      // 4096 B  (layer1/3 output, row stride 128)
    float qsc [64 * 16];       // 4096 B  query f32 [head][k]
    float rmax[64];
    float rinv[64];            // reciprocal of row sum
};                             // 13824 B

__global__ void __launch_bounds__(64) interp_main(
    const float* __restrict__ pos, const float* __restrict__ pos_nm,
    const int*   __restrict__ proj, const u16* __restrict__ latT,
    const u16* __restrict__ W1f, const u16* __restrict__ W2f, const u16* __restrict__ W3f,
    const u16* __restrict__ Wqf, const u16* __restrict__ Wvf,
    const float* __restrict__ b1, const float* __restrict__ b2, const float* __restrict__ b3,
    const float* __restrict__ bq, const float* __restrict__ bv,
    const float* __restrict__ w8, const float* __restrict__ bi8,
    float* __restrict__ out)
{
    __shared__ __align__(16) PtLds lds[2][2];          // [wave][point] = 55296 B
    const int wave = threadIdx.x >> 5, lane = threadIdx.x & 31;
    PtLds& L0 = lds[wave][0];
    PtLds& L1 = lds[wave][1];
    const int tile0 = blockIdx.x * 4 + wave * 2;
    const int col = lane & 15, lg = lane >> 4;

    // ---- gather both points: each lane loads 32 contiguous bf16 channels of one neighbor ----
#pragma unroll
    for (int p = 0; p < 2; ++p) {
        PtLds& L = lds[wave][p];
        const int tile = tile0 + p;
        const int b = tile >> 15;                      // M = 2^15
        const int m = tile & (MPTS - 1);
        const int k = lane >> 1, h = lane & 1;
        const int idx = proj[((size_t)b * MPTS + m) * KNN + k];
        const uint4* src = reinterpret_cast<const uint4*>(latT + ((size_t)b * NPTS + idx) * 128 + h * 64);
        uint4 r0 = src[0], r1 = src[1], r2 = src[2], r3 = src[3];
        uint4* dst = reinterpret_cast<uint4*>(L.bufA + k * IC1PAD + h * 64);
        dst[0] = r0; dst[1] = r1; dst[2] = r2; dst[3] = r3;
        if (h == 0) {   // relative coords into padded channels 128..159
            float d0 = pos_nm[((size_t)b * 3 + 0) * MPTS + m] - pos[((size_t)b * 3 + 0) * NPTS + idx];
            float d1 = pos_nm[((size_t)b * 3 + 1) * MPTS + m] - pos[((size_t)b * 3 + 1) * NPTS + idx];
            float d2 = pos_nm[((size_t)b * 3 + 2) * MPTS + m] - pos[((size_t)b * 3 + 2) * NPTS + idx];
            uint4 z0, zz = {0u, 0u, 0u, 0u};
            z0.x = (unsigned)f2bf(d0) | ((unsigned)f2bf(d1) << 16);
            z0.y = (unsigned)f2bf(d2); z0.z = 0u; z0.w = 0u;
            uint4* pdst = reinterpret_cast<uint4*>(L.bufA + k * IC1PAD + 128);
            pdst[0] = z0; pdst[1] = zz; pdst[2] = zz; pdst[3] = zz;
        }
    }
    asm volatile("s_wait_dscnt 0x0" ::: "memory");

    // ---- layer 1: bufA(160, 5 chunks) -> bufB, relu ----
    for (int rt = 0; rt < 8; ++rt) {
        v8f a0 = {}, a1 = {};
        gemm_tile2(W1f, IC1PAD, rt, L0.bufA, L1.bufA, IC1PAD, lane, 5, a0, a1);
        const int base = rt * 16 + 8 * lg;
        epi_relu_store(a0, b1, base, L0.bufB + col * 128 + base);
        epi_relu_store(a1, b1, base, L1.bufB + col * 128 + base);
    }
    asm volatile("s_wait_dscnt 0x0" ::: "memory");

    // ---- layer 2: bufB -> bufA (row stride 160, first 128 ch), relu ----
    for (int rt = 0; rt < 8; ++rt) {
        v8f a0 = {}, a1 = {};
        gemm_tile2(W2f, 128, rt, L0.bufB, L1.bufB, 128, lane, 4, a0, a1);
        const int base = rt * 16 + 8 * lg;
        epi_relu_store(a0, b2, base, L0.bufA + col * IC1PAD + base);
        epi_relu_store(a1, b2, base, L1.bufA + col * IC1PAD + base);
    }
    asm volatile("s_wait_dscnt 0x0" ::: "memory");

    // ---- layer 3: bufA(stride 160, 4 chunks) -> bufB, relu ----
    for (int rt = 0; rt < 8; ++rt) {
        v8f a0 = {}, a1 = {};
        gemm_tile2(W3f, 128, rt, L0.bufA, L1.bufA, IC1PAD, lane, 4, a0, a1);
        const int base = rt * 16 + 8 * lg;
        epi_relu_store(a0, b3, base, L0.bufB + col * 128 + base);
        epi_relu_store(a1, b3, base, L1.bufB + col * 128 + base);
    }
    asm volatile("s_wait_dscnt 0x0" ::: "memory");

    // ---- query: 64x16 f32 into qsc (both points) ----
    for (int rt = 0; rt < 4; ++rt) {
        v8f a0 = {}, a1 = {};
        gemm_tile2(Wqf, 128, rt, L0.bufB, L1.bufB, 128, lane, 4, a0, a1);
        const int base = rt * 16 + 8 * lg;
#pragma unroll
        for (int v = 0; v < 8; ++v) {
            float bb = bq[base + v];
            L0.qsc[(base + v) * 16 + col] = a0[v] + bb;
            L1.qsc[(base + v) * 16 + col] = a1[v] + bb;
        }
    }
    asm volatile("s_wait_dscnt 0x0" ::: "memory");

    // ---- softmax over K per head: pass1 row max + reciprocal-sum (2 rows/lane/point) ----
#pragma unroll
    for (int p = 0; p < 2; ++p) {
        PtLds& L = lds[wave][p];
        for (int rr = 0; rr < 2; ++rr) {
            const int r = lane * 2 + rr;
            float x[16];
#pragma unroll
            for (int i = 0; i < 16; ++i) x[i] = L.qsc[r * 16 + i];
            float mx = x[0];
#pragma unroll
            for (int i = 1; i < 16; ++i) mx = fmaxf(mx, x[i]);
            float sm = 0.f;
#pragma unroll
            for (int i = 0; i < 16; ++i) sm += __expf(x[i] - mx);
            L.rmax[r] = mx; L.rinv[r] = 1.f / sm;
        }
    }
    asm volatile("s_wait_dscnt 0x0" ::: "memory");

    // ---- pass2: attention per column (mean over 64 heads), per point ----
    float at[2];
#pragma unroll
    for (int p = 0; p < 2; ++p) {
        PtLds& L = lds[wave][p];
        float a = 0.f;
        for (int r = lg * 32; r < lg * 32 + 32; ++r)
            a += __expf(L.qsc[r * 16 + col] - L.rmax[r]) * L.rinv[r];
        a += __shfl_xor(a, 16);
        at[p] = a * (1.f / 64.f);
    }

    // ---- value WMMAs fused with aggregation and final 128->2 projection ----
    float o0[2] = {0.f, 0.f}, o1[2] = {0.f, 0.f};
    for (int rt = 0; rt < 8; ++rt) {
        v8f a0 = {}, a1 = {};
        gemm_tile2(Wvf, 128, rt, L0.bufB, L1.bufB, 128, lane, 4, a0, a1);
        const int base = rt * 16 + 8 * lg;
#pragma unroll
        for (int v = 0; v < 8; ++v) {
            const int ch = base + v;
            const float bb = bv[ch], w0 = w8[ch], w1 = w8[128 + ch];
            float f0 = (a0[v] + bb) * at[0];
            float f1 = (a1[v] + bb) * at[1];
            o0[0] += w0 * f0; o1[0] += w1 * f0;
            o0[1] += w0 * f1; o1[1] += w1 * f1;
        }
    }
    // reduce over all 32 lanes (covers all (ch, k) pairs exactly once)
#pragma unroll
    for (int p = 0; p < 2; ++p) {
        float a = o0[p], bsum = o1[p];
        a += __shfl_xor(a, 1); a += __shfl_xor(a, 2); a += __shfl_xor(a, 4);
        a += __shfl_xor(a, 8); a += __shfl_xor(a, 16);
        bsum += __shfl_xor(bsum, 1); bsum += __shfl_xor(bsum, 2); bsum += __shfl_xor(bsum, 4);
        bsum += __shfl_xor(bsum, 8); bsum += __shfl_xor(bsum, 16);
        if (lane == 0) {
            const int tile = tile0 + p;
            const int b = tile >> 15;
            const int m = tile & (MPTS - 1);
            out[((size_t)b * 2 + 0) * MPTS + m] = a + bi8[0];
            out[((size_t)b * 2 + 1) * MPTS + m] = bsum + bi8[1];
        }
    }
}

// ---------- workspace layout (bytes) ----------
#define OFF_W1   0u          // 128*160*2 = 40960
#define OFF_W2   40960u      // 32768
#define OFF_W3   73728u      // 32768
#define OFF_WQ   106496u     // 16384
#define OFF_WV   122880u     // 32768
#define OFF_B1   155648u     // 512
#define OFF_B2   156160u
#define OFF_B3   156672u
#define OFF_BQ   157184u     // 256
#define OFF_BV   157440u     // 512
#define OFF_LATT 163840u     // 2*16384*128*2 = 8388608  (total ~8.56 MB)

extern "C" void kernel_launch(void* const* d_in, const int* in_sizes, int n_in,
                              void* d_out, int out_size, void* d_ws, size_t ws_size,
                              hipStream_t stream) {
    const float* latents = (const float*)d_in[0];
    const float* pos     = (const float*)d_in[1];
    const float* pos_nm  = (const float*)d_in[2];
    const int*   proj    = (const int*)  d_in[3];
    const float* P[32];
    for (int i = 0; i < 32; ++i) P[i] = (const float*)d_in[4 + i];
    // P: w1 bi1 g1 be1 m1 v1 | w2.. | w3.. | wq.. | wv.. | w8 bi8

    unsigned char* ws = (unsigned char*)d_ws;
    u16*   W1f = (u16*)  (ws + OFF_W1);
    u16*   W2f = (u16*)  (ws + OFF_W2);
    u16*   W3f = (u16*)  (ws + OFF_W3);
    u16*   Wqf = (u16*)  (ws + OFF_WQ);
    u16*   Wvf = (u16*)  (ws + OFF_WV);
    float* b1  = (float*)(ws + OFF_B1);
    float* b2  = (float*)(ws + OFF_B2);
    float* b3  = (float*)(ws + OFF_B3);
    float* bq  = (float*)(ws + OFF_BQ);
    float* bv  = (float*)(ws + OFF_BV);
    u16*   latT= (u16*)  (ws + OFF_LATT);

    prep_weights<<<5, 128, 0, stream>>>(
        P[0], P[1], P[2], P[3], P[4], P[5],
        P[6], P[7], P[8], P[9], P[10], P[11],
        P[12], P[13], P[14], P[15], P[16], P[17],
        P[18], P[19], P[20], P[21], P[22], P[23],
        P[24], P[25], P[26], P[27], P[28], P[29],
        W1f, W2f, W3f, Wqf, Wvf, b1, b2, b3, bq, bv);

    const int totT = 2 * NPTS * 128;
    prep_latT<<<(totT + 255) / 256, 256, 0, stream>>>(latents, latT, totT);

    interp_main<<<(2 * MPTS) / 4, 64, 0, stream>>>(
        pos, pos_nm, proj, latT, W1f, W2f, W3f, Wqf, Wvf,
        b1, b2, b3, bq, bv, P[30], P[31], (float*)d_out);
}
